// FeatDecoder_82454782149367
// MI455X (gfx1250) — matrix operand order repaired and checked by reading, not compiled, not run on hardware
//
#include <hip/hip_runtime.h>

typedef __attribute__((ext_vector_type(16))) _Float16 v16h;
typedef __attribute__((ext_vector_type(8)))  float    v8f;

// ---------------- zero workspace ----------------
__global__ __launch_bounds__(256) void fd_zero(float* __restrict__ p, int n) {
    int i = blockIdx.x * blockDim.x + threadIdx.x;
    if (i < n) p[i] = 0.0f;
}

// Build the A-fragment for one 16x32 f16 tile (contiguous k order, wave32):
// lane (m = lane%16, hi = lane/16): halves[j]   = FC[m][kbase + 16*hi + j]
//                                   halves[8+j] = FC[m][kbase + 16*hi + 8 + j]
// `base` points at FC row + kbase; `scale` zeroes the aux-miss case without divergence.
__device__ __forceinline__ v16h cvt_a(const float* __restrict__ base, int hi, float scale) {
    const float4* p0 = (const float4*)(base + 16 * hi);
    const float4* p1 = (const float4*)(base + 16 * hi + 8);
    float4 x0 = p0[0], x1 = p0[1];
    float4 y0 = p1[0], y1 = p1[1];
    v16h a;
    a[0]  = (_Float16)(x0.x * scale); a[1]  = (_Float16)(x0.y * scale);
    a[2]  = (_Float16)(x0.z * scale); a[3]  = (_Float16)(x0.w * scale);
    a[4]  = (_Float16)(x1.x * scale); a[5]  = (_Float16)(x1.y * scale);
    a[6]  = (_Float16)(x1.z * scale); a[7]  = (_Float16)(x1.w * scale);
    a[8]  = (_Float16)(y0.x * scale); a[9]  = (_Float16)(y0.y * scale);
    a[10] = (_Float16)(y0.z * scale); a[11] = (_Float16)(y0.w * scale);
    a[12] = (_Float16)(y1.x * scale); a[13] = (_Float16)(y1.y * scale);
    a[14] = (_Float16)(y1.z * scale); a[15] = (_Float16)(y1.w * scale);
    return a;
}

// B-fragment (32x16 f16, contiguous k order, as in A): lane (N = lane%16, hi = lane/16)
// holds halves[j] = B[K = 16*hi + j][N]. We replicate the w-chunk across all
// 16 columns so every column of D equals the dot-product vector.
// `wbase` points at w + kbase + 16*hi.
__device__ __forceinline__ v16h cvt_b(const float* __restrict__ wbase) {
    const float4* p = (const float4*)wbase;
    float4 c0 = p[0], c1 = p[1], c2 = p[2], c3 = p[3];
    v16h b;
    b[0]  = (_Float16)c0.x; b[1]  = (_Float16)c0.y;
    b[2]  = (_Float16)c0.z; b[3]  = (_Float16)c0.w;
    b[4]  = (_Float16)c1.x; b[5]  = (_Float16)c1.y;
    b[6]  = (_Float16)c1.z; b[7]  = (_Float16)c1.w;
    b[8]  = (_Float16)c2.x; b[9]  = (_Float16)c2.y;
    b[10] = (_Float16)c2.z; b[11] = (_Float16)c2.w;
    b[12] = (_Float16)c3.x; b[13] = (_Float16)c3.y;
    b[14] = (_Float16)c3.z; b[15] = (_Float16)c3.w;
    return b;
}

// One wave32 handles 16 points: fc[16 x 96] . w[96] via three
// v_wmma_f32_16x16x32_f16 (fp32 accumulate), then scalar scatter-atomics.
__global__ __launch_bounds__(256) void fd_dot_scatter(
    const float* __restrict__ feats,        // [N,64]
    const float* __restrict__ aux,          // [M,32]
    const float* __restrict__ w,            // [96]
    const int*   __restrict__ idx_map,      // [N]
    const int*   __restrict__ reduce_idx,   // [N]
    float* __restrict__ ssum,               // [K]
    float* __restrict__ cnt,                // [K]
    int n, int numTiles)
{
    const int gwave = (int)((blockIdx.x * blockDim.x + threadIdx.x) >> 5);
    const int lane  = (int)(threadIdx.x & 31);
    if (gwave >= numTiles) return;          // wave-uniform: EXEC all-ones below

    const int row = lane & 15;              // M index this lane feeds
    const int hi  = lane >> 4;              // half-wave selector
    const int p   = gwave * 16 + row;       // point owned by this A-row
    const bool valid = (p < n);
    const int pc  = valid ? p : (n - 1);    // clamp tail loads (rows unscattered)

    // B fragments: w chunks for K-tiles [0,32), [32,64), [64,96)
    v16h b0 = cvt_b(w + 0  + 16 * hi);
    v16h b1 = cvt_b(w + 32 + 16 * hi);
    v16h b2 = cvt_b(w + 64 + 16 * hi);

    // A fragments: feats channels 0-31, 32-63; gathered aux channels 64-95
    const float* frow = feats + (long)pc * 64;
    v16h a0 = cvt_a(frow,      hi, 1.0f);
    v16h a1 = cvt_a(frow + 32, hi, 1.0f);

    const int aidx  = idx_map[pc];
    const float msk = (aidx >= 0) ? 1.0f : 0.0f;            // -1 => zeros
    const float* arow = aux + (long)(aidx >= 0 ? aidx : 0) * 32;
    v16h a2 = cvt_a(arow, hi, msk);

    v8f acc = {};
    acc = __builtin_amdgcn_wmma_f32_16x16x32_f16(false, a0, false, b0,
                                                 (short)0, acc, false, false);
    acc = __builtin_amdgcn_wmma_f32_16x16x32_f16(false, a1, false, b1,
                                                 (short)0, acc, false, false);
    acc = __builtin_amdgcn_wmma_f32_16x16x32_f16(false, a2, false, b2,
                                                 (short)0, acc, false, false);

    // Segment counts: one lane per point.
    if (hi == 0 && valid) {
        atomicAdd(&cnt[reduce_idx[p]], 1.0f);
    }

    // D column N=0: lane 0 holds M=0..7 (VGPR r -> M=r),
    //               lane 16 holds M=8..15 (VGPR r -> M=8+r).
    if (row == 0) {
        const int base = gwave * 16 + hi * 8;
        #pragma unroll
        for (int r = 0; r < 8; ++r) {
            const int pp = base + r;
            if (pp < n) {
                atomicAdd(&ssum[reduce_idx[pp]], acc[r]);
            }
        }
    }
}

// out[k] = ssum[k] / max(cnt[k], 1) + bias
__global__ __launch_bounds__(256) void fd_finalize(
    const float* __restrict__ ssum, const float* __restrict__ cnt,
    const float* __restrict__ bias, float* __restrict__ out, int k)
{
    int i = blockIdx.x * blockDim.x + threadIdx.x;
    if (i < k) {
        out[i] = ssum[i] / fmaxf(cnt[i], 1.0f) + bias[0];
    }
}

extern "C" void kernel_launch(void* const* d_in, const int* in_sizes, int n_in,
                              void* d_out, int out_size, void* d_ws, size_t ws_size,
                              hipStream_t stream) {
    (void)n_in; (void)ws_size;
    const float* feats      = (const float*)d_in[0];   // [N,64]
    const float* aux_feats  = (const float*)d_in[1];   // [M,32]
    const float* proj_w     = (const float*)d_in[2];   // [96,1]
    const float* proj_b     = (const float*)d_in[3];   // [1]
    const int*   idx_map    = (const int*)d_in[4];     // [N]
    const int*   reduce_idx = (const int*)d_in[5];     // [N]
    float* out = (float*)d_out;                        // [K,1] fp32

    const int n = in_sizes[4];        // N
    const int k = out_size;           // K (OUT == 1)

    float* ssum = (float*)d_ws;       // [K]
    float* cnt  = ssum + k;           // [K]

    // 1) zero accumulators (every launch: graph replays don't re-poison)
    const int nz = 2 * k;
    fd_zero<<<(nz + 255) / 256, 256, 0, stream>>>(ssum, nz);

    // 2) fused gather + f16-WMMA dot + scalar scatter-add
    const int numTiles = (n + 15) / 16;            // one wave32 per 16 points
    const int blocks   = (numTiles + 7) / 8;       // 8 waves / 256-thread block
    fd_dot_scatter<<<blocks, 256, 0, stream>>>(feats, aux_feats, proj_w,
                                               idx_map, reduce_idx,
                                               ssum, cnt, n, numTiles);

    // 3) divide by counts, add bias
    fd_finalize<<<(k + 255) / 256, 256, 0, stream>>>(ssum, cnt, proj_b, out, k);
}
